// LightGCN_62405874811873
// MI455X (gfx1250) — matrix operand order repaired and checked by reading, not compiled, not run on hardware
//
#include <hip/hip_runtime.h>

#define N_USERS  100000
#define N_ITEMS  200000
#define N_NODES  (N_USERS + N_ITEMS)
#define D        64
#define BATCH    4096
#define N_LAYERS 3

typedef __attribute__((ext_vector_type(16))) _Float16 v16h;
typedef __attribute__((ext_vector_type(8)))  float    v8f;

// cur = concat(U_emb, V_emb)
__global__ __launch_bounds__(256) void concat_emb(const float* __restrict__ U,
                                                  const float* __restrict__ V,
                                                  float* __restrict__ cur) {
    long idx = (long)blockIdx.x * blockDim.x + threadIdx.x;
    const long nU = (long)N_USERS * D;
    const long nT = (long)N_NODES * D;
    if (idx < nT) cur[idx] = (idx < nU) ? U[idx] : V[idx - nU];
}

// Fu[b,:] += src[u[b],:];  Fi[b,:] += src[N_USERS+i[b],:]
__global__ __launch_bounds__(256) void gather_add(const int* __restrict__ u,
                                                  const int* __restrict__ it,
                                                  const float* __restrict__ src,
                                                  float* __restrict__ Fu,
                                                  float* __restrict__ Fi) {
    int t = blockIdx.x * blockDim.x + threadIdx.x;   // 0 .. BATCH*D-1
    if (t >= BATCH * D) return;
    int b = t >> 6;
    int d = t & 63;
    long un = (long)u[b];
    long in = (long)it[b] + N_USERS;
    Fu[t] += src[un * D + d];
    Fi[t] += src[in * D + d];
}

// One wave per edge: lanes cover 64 dims as float2 (coalesced 256B gather +
// 256B contiguous atomic burst). Edge list read non-temporally so the 120MB/layer
// stream does not evict the 76.8MB embedding table resident in the 192MB L2.
__global__ __launch_bounds__(256) void spmm_scatter(const int*   __restrict__ rows,
                                                    const int*   __restrict__ cols,
                                                    const float* __restrict__ vals,
                                                    const float* __restrict__ cur,
                                                    float*       __restrict__ nxt,
                                                    int nEdges) {
    int lane = threadIdx.x & 31;
    int edge = blockIdx.x * (blockDim.x >> 5) + (threadIdx.x >> 5);
    if (edge >= nEdges) return;

    int   r = __builtin_nontemporal_load(rows + edge);
    int   c = __builtin_nontemporal_load(cols + edge);
    float v = __builtin_nontemporal_load(vals + edge);

    const float2* src = (const float2*)(cur + (long)c * D) + lane;
    float2 x = *src;                               // L2-resident gather

    float* dst = nxt + (long)r * D + lane * 2;
    unsafeAtomicAdd(dst,     v * x.x);             // hw global_atomic_add_f32
    unsafeAtomicAdd(dst + 1, v * x.y);
}

// Scores for one tile of 16 batch pairs = diag( Fu_tile(16x64) * Fi_tile^T(64x16) )
// via 2x V_WMMA_F32_16X16X32_F16. One wave per tile (EXEC all-ones at WMMA).
__global__ __launch_bounds__(32) void score_wmma(const float* __restrict__ Fu,
                                                 const float* __restrict__ Fi,
                                                 float* __restrict__ out) {
    int tile = blockIdx.x;       // BATCH/16 tiles
    int lane = threadIdx.x;      // 0..31
    int half = lane >> 4;        // lane group
    int m    = lane & 15;        // A-row for this lane / B-col for this lane

    const float* fu = Fu + ((long)tile * 16 + m) * D;
    const float* fi = Fi + ((long)tile * 16 + m) * D;

    v8f c = {};
#pragma unroll
    for (int chunk = 0; chunk < 2; ++chunk) {       // K = 64 = 2 x 32
        int kc = chunk * 32;
        v16h a, b;
        // A (16x32 f16): VGPR v<4 holds K = half*8 + 2v,2v+1; v>=4 adds +16.
#pragma unroll
        for (int v = 0; v < 4; ++v) {
            int k = kc + half * 8 + 2 * v;
            a[2 * v]           = (_Float16)fu[k];
            a[2 * v + 1]       = (_Float16)fu[k + 1];
            a[2 * (v + 4)]     = (_Float16)fu[k + 16];
            a[2 * (v + 4) + 1] = (_Float16)fu[k + 17];
        }
        // B (32x16 f16): same K order as A: VGPR v<4 holds K = half*8 + 2v,2v+1; v>=4 adds +16.
#pragma unroll
        for (int v = 0; v < 8; ++v) {
            int k = kc + half * 8 + 2 * (v & 3) + (v >> 2) * 16;
            b[2 * v]     = (_Float16)fi[k];
            b[2 * v + 1] = (_Float16)fi[k + 1];
        }
        c = __builtin_amdgcn_wmma_f32_16x16x32_f16(
                /*neg_a=*/false, a, /*neg_b=*/false, b,
                /*c_mod=*/(short)0, c, /*reuse_a=*/false, /*reuse_b=*/false);
    }

    // Diagonal of 16x16 f32 C: M=m lives in lane m (m<8, VGPR m) or lane m+16
    // (m>=8, VGPR m-8). So lanes 0-7 take c[lane], lanes 24-31 take c[lane-24].
    int r = (lane < 8) ? lane : lane - 24;
    float s = c[0];
    if (r == 1) s = c[1];
    if (r == 2) s = c[2];
    if (r == 3) s = c[3];
    if (r == 4) s = c[4];
    if (r == 5) s = c[5];
    if (r == 6) s = c[6];
    if (r == 7) s = c[7];
    if (lane < 8 || lane >= 24) {
        int mm = (lane < 8) ? lane : lane - 16;
        out[tile * 16 + mm] = s * 0.0625f;   // (1/4)*(1/4) layer-mean scaling
    }
}

extern "C" void kernel_launch(void* const* d_in, const int* in_sizes, int n_in,
                              void* d_out, int out_size, void* d_ws, size_t ws_size,
                              hipStream_t stream) {
    const int*   u    = (const int*)d_in[0];
    const int*   it   = (const int*)d_in[1];
    const int*   rows = (const int*)d_in[2];
    const int*   cols = (const int*)d_in[3];
    const float* vals = (const float*)d_in[4];
    const float* U    = (const float*)d_in[5];
    const float* V    = (const float*)d_in[6];
    const int nEdges  = in_sizes[2];

    const size_t nd = (size_t)N_NODES * D;            // 19.2M floats
    float* bufA = (float*)d_ws;                       // ping
    float* bufB = bufA + nd;                          // pong
    float* Fu   = bufB + nd;                          // BATCH x D
    float* Fi   = Fu + (size_t)BATCH * D;

    // E0 = concat(U,V); Fu/Fi = layer-0 contribution
    concat_emb<<<(unsigned)((nd + 255) / 256), 256, 0, stream>>>(U, V, bufA);
    hipMemsetAsync(Fu, 0, 2 * (size_t)BATCH * D * sizeof(float), stream);
    gather_add<<<(BATCH * D) / 256, 256, 0, stream>>>(u, it, bufA, Fu, Fi);

    float* cur = bufA;
    float* nxt = bufB;
    for (int layer = 0; layer < N_LAYERS; ++layer) {
        hipMemsetAsync(nxt, 0, nd * sizeof(float), stream);
        int nBlocks = (nEdges + 7) / 8;               // 8 waves (edges) per block
        spmm_scatter<<<nBlocks, 256, 0, stream>>>(rows, cols, vals, cur, nxt, nEdges);
        gather_add<<<(BATCH * D) / 256, 256, 0, stream>>>(u, it, nxt, Fu, Fi);
        float* t = cur; cur = nxt; nxt = t;           // host-side deterministic swap
    }

    score_wmma<<<BATCH / 16, 32, 0, stream>>>(Fu, Fi, (float*)d_out);
}
